// KolmogorovArnoldNetwork_45406394253366
// MI455X (gfx1250) — compile-verified, matrix operand order checked by reading, not run on hardware
//
#include <hip/hip_runtime.h>
#include <hip/hip_bf16.h>

typedef __attribute__((ext_vector_type(16))) _Float16 v16h;
typedef __attribute__((ext_vector_type(8)))  float    v8f;

#define HID   256
#define CCH   16
#define WAVES 8
#define BLOCK (WAVES * 32)
#define SLABS 4
#define ROWS_PER_WG (WAVES * 16 * SLABS)   // 512 rows per workgroup

// LDS layout:
//   [0, 128KB)  : W2[c] as f16, pre-swizzled into per-lane WMMA B fragments.
//                 Fragment f = kc*512 + t*32 + lane holds the 16 halves lane
//                 `lane` needs for the WMMA on K-chunk kc, N-tile t.
//   then        : W1[c], b1[c], b2[c] (256 f32 each), W3[c] (768 f32)
#define W2_HALVES (8 * 16 * 32 * 16)                 // 65536 halves = 128 KB
#define SMEM_BYTES (W2_HALVES * 2 + (256 * 3 + 768) * 4)

__global__ __launch_bounds__(BLOCK) void kan_fused_wmma_kernel(
    const float* __restrict__ xg,  const float* __restrict__ W1g,
    const float* __restrict__ b1g, const float* __restrict__ W2g,
    const float* __restrict__ b2g, const float* __restrict__ W3g,
    const float* __restrict__ b3g, float* __restrict__ outg)
{
    extern __shared__ char smem[];
    _Float16* w2s = (_Float16*)smem;
    float*    w1s = (float*)(smem + W2_HALVES * 2);
    float*    b1s = w1s + 256;
    float*    b2s = b1s + 256;
    float*    w3s = b2s + 256;   // 768 floats, row-major [n][o]

    const int c    = blockIdx.y;
    const int tid  = threadIdx.x;
    const int lane = tid & 31;
    const int wave = tid >> 5;
    const int q    = lane & 15;   // column-in-tile / row selector
    const int hi   = lane >> 4;   // lane half (WMMA layout group)

    // ---- stage small per-channel tensors ----
    for (int i = tid; i < 256; i += BLOCK) {
        w1s[i] = W1g[c * HID + i];
        b1s[i] = b1g[c * HID + i];
        b2s[i] = b2g[c * HID + i];
    }
    for (int i = tid; i < 768; i += BLOCK)
        w3s[i] = W3g[c * HID * 3 + i];

    // ---- stage W2[c] (f32 [K=256][N=256]) -> swizzled f16 B fragments ----
    // 16-bit B 32x16 layout: lanes 0-15 (N=lane): halves 0..7 -> K 0..7,
    // halves 8..15 -> K 16..23; lanes 16-31: halves 0..7 -> K 8..15,
    // halves 8..15 -> K 24..31 (within the 32-wide K chunk).
    const float* W2c = W2g + (size_t)c * HID * HID;
#pragma unroll 1
    for (int f = tid; f < 8 * 16 * 32; f += BLOCK) {
        const int l  = f & 31;
        const int t  = (f >> 5) & 15;
        const int kc = f >> 9;
        const int n  = t * 16 + (l & 15);
        const int kb = kc * 32 + ((l & 16) ? 8 : 0);
        v16h frag;
#pragma unroll
        for (int h = 0; h < 16; ++h) {
            const int k = kb + ((h < 8) ? h : (h + 8));
            frag[h] = (_Float16)W2c[k * HID + n];
        }
        *(v16h*)(w2s + (size_t)f * 16) = frag;   // 32B aligned, conflict-free
    }
    __syncthreads();

    const size_t rowBase0 =
        (size_t)blockIdx.x * ROWS_PER_WG + (size_t)wave * (16 * SLABS);
    const float bo0 = b3g[c * 3 + 0];
    const float bo1 = b3g[c * 3 + 1];
    const float bo2 = b3g[c * 3 + 2];

    // Keep the slab loop rolled: each iteration's steady-state live set is
    // ~110 VGPRs (A slab 64 + acc 8 + B frag 8 + fc3 partials 24). Full
    // unrolling previously pushed allocation past v255 and into scratch.
#pragma unroll 1
    for (int slab = 0; slab < SLABS; ++slab) {
        const size_t rowBase = rowBase0 + (size_t)slab * 16;

        // ---- fc1 fused into A fragments: A[m,k] = relu(x_m*W1[k]+b1[k]) ----
        // A 16x32 layout mirrors B: row m = lane&15; K mapping as above.
        const float xm = xg[(rowBase + q) * CCH + c];
        v16h a[8];
#pragma unroll
        for (int kc = 0; kc < 8; ++kc) {
            const int kb = kc * 32 + hi * 8;
#pragma unroll
            for (int h = 0; h < 16; ++h) {
                const int k = kb + ((h < 8) ? h : (h + 8));
                float v = fmaf(xm, w1s[k], b1s[k]);
                a[kc][h] = (_Float16)(v > 0.f ? v : 0.f);
            }
        }

        // ---- fc2 via WMMA, fc3 folded onto the D layout ----
        float p0[8], p1[8], p2[8];
#pragma unroll
        for (int j = 0; j < 8; ++j) { p0[j] = 0.f; p1[j] = 0.f; p2[j] = 0.f; }

        // Rolled: 8 back-to-back WMMAs + interleaved LDS B loads per trip.
#pragma unroll 1
        for (int t = 0; t < 16; ++t) {           // 16 N-tiles of fc2 output
            v8f acc = {};
#pragma unroll
            for (int kc = 0; kc < 8; ++kc) {     // K = 256 in 32-chunks
                const v16h b = *(const v16h*)(w2s + ((kc * 512 + t * 32 + lane) << 4));
                acc = __builtin_amdgcn_wmma_f32_16x16x32_f16(
                    false, a[kc], false, b, (short)0, acc, false, false);
            }
            // D layout: VGPR j of this lane = element (M = j + 8*hi, N = t*16+q)
            const int   n  = t * 16 + q;
            const float bb = b2s[n];
            const float u0 = w3s[n * 3 + 0];
            const float u1 = w3s[n * 3 + 1];
            const float u2 = w3s[n * 3 + 2];
#pragma unroll
            for (int j = 0; j < 8; ++j) {
                float v = acc[j] + bb;
                v = v > 0.f ? v : 0.f;           // relu(fc2 + b2)
                p0[j] = fmaf(v, u0, p0[j]);
                p1[j] = fmaf(v, u1, p1[j]);
                p2[j] = fmaf(v, u2, p2[j]);
            }
        }

        // ---- reduce fc3 partials across the 16 lanes of each half ----
#pragma unroll
        for (int j = 0; j < 8; ++j) {
#pragma unroll
            for (int m = 1; m <= 8; m <<= 1) {
                p0[j] += __shfl_xor(p0[j], m, 32);
                p1[j] += __shfl_xor(p1[j], m, 32);
                p2[j] += __shfl_xor(p2[j], m, 32);
            }
        }

        // ---- write out[row, c*3 + o]; lane q==j writes row j + 8*hi ----
#pragma unroll
        for (int j = 0; j < 8; ++j) {
            if (q == j) {
                const size_t row = rowBase + (size_t)j + (size_t)(8 * hi);
                float* op = outg + row * (CCH * 3) + c * 3;
                op[0] = p0[j] + bo0;
                op[1] = p1[j] + bo1;
                op[2] = p2[j] + bo2;
            }
        }
    }
}

extern "C" void kernel_launch(void* const* d_in, const int* in_sizes, int n_in,
                              void* d_out, int out_size, void* d_ws, size_t ws_size,
                              hipStream_t stream) {
    (void)in_sizes; (void)n_in; (void)d_ws; (void)ws_size; (void)out_size;
    const float* x  = (const float*)d_in[0];
    const float* W1 = (const float*)d_in[1];
    const float* b1 = (const float*)d_in[2];
    const float* W2 = (const float*)d_in[3];
    const float* b2 = (const float*)d_in[4];
    const float* W3 = (const float*)d_in[5];
    const float* b3 = (const float*)d_in[6];
    float* out = (float*)d_out;

    // 137 KB dynamic LDS (> default 64 KB cap) — raise the attribute.
    hipFuncSetAttribute((const void*)kan_fused_wmma_kernel,
                        hipFuncAttributeMaxDynamicSharedMemorySize, SMEM_BYTES);

    const int Nrows = 8 * 64 * 64;                       // 32768
    dim3 grid(Nrows / ROWS_PER_WG, CCH);                 // (64, 16)
    kan_fused_wmma_kernel<<<grid, BLOCK, SMEM_BYTES, stream>>>(
        x, W1, b1, W2, b2, W3, b3, out);
}
